// ANFM_36318243455110
// MI455X (gfx1250) — compile-verified, matrix operand order checked by reading, not run on hardware
//
#include <hip/hip_runtime.h>
#include <hip/hip_bf16.h>

// ---------------------------------------------------------------------------
// ANFM fused pipeline for gfx1250 (MI455X), bf16 WMMA path.
//   kernel 0: pack fp32 weights -> bf16 B-fragment tiles (workspace)
//   kernel 1: gather emb -> silu(x@W1^T+b1) -> @W2^T+b2 = h (bf16, workspace)
//   kernel 2: pair scores (att GEMM + dots + softmax) and pair MLP, reduce.
// All GEMMs on v_wmma_f32_16x16x32_bf16; staging vectorized to b128 and the
// kernel-2 row staging uses GLOBAL_LOAD_ASYNC_TO_LDS_B128 (ASYNCcnt).
// ---------------------------------------------------------------------------

typedef __attribute__((ext_vector_type(16))) __bf16 v16bf;
typedef __attribute__((ext_vector_type(8)))  float  v8f;

#define TEAM 5
#define NPAIR 20          // TEAM*(TEAM-1)
#define B_TOTAL 16384
#define ROWS_TOTAL (B_TOTAL * TEAM)   // 81920
#define PD 256
#define HD 256
#define H2 512
#define MLP_H 512

// ---- bf16 helpers (round-to-nearest-even) ---------------------------------
__device__ inline unsigned short f2bfbits(float f) {
  union { float f; unsigned u; } v; v.f = f;
  unsigned r = (v.u + 0x7FFFu + ((v.u >> 16) & 1u)) >> 16;
  return (unsigned short)r;
}
__device__ inline float bf2f(unsigned short s) {
  union { unsigned u; float f; } o; o.u = ((unsigned)s) << 16; return o.f;
}
// packed helpers on 2x bf16 in one dword
__device__ inline unsigned pkmul_bf16(unsigned a, unsigned b) {
  float lo = bf2f((unsigned short)(a & 0xFFFFu)) * bf2f((unsigned short)(b & 0xFFFFu));
  float hi = bf2f((unsigned short)(a >> 16))     * bf2f((unsigned short)(b >> 16));
  return (unsigned)f2bfbits(lo) | ((unsigned)f2bfbits(hi) << 16);
}
__device__ inline float dot2_acc(unsigned a, unsigned b, float s) {
  s += bf2f((unsigned short)(a & 0xFFFFu)) * bf2f((unsigned short)(b & 0xFFFFu));
  s += bf2f((unsigned short)(a >> 16))     * bf2f((unsigned short)(b >> 16));
  return s;
}

// ---- fragment loaders ------------------------------------------------------
// A-matrix 16x32 bf16 layout (wave32): lane&15 = row M; lanes<16 hold
// K = {kt*32+0..7, kt*32+16..23}, lanes>=16 hold K = {+8..15, +24..31}.
__device__ inline v16bf load_a_lds(const unsigned short* lds, int ld, int kt, int lane) {
  int m  = lane & 15;
  int kb = kt * 32 + ((lane & 16) ? 8 : 0);
  const uint4* p = (const uint4*)(lds + m * ld + kb);
  union { uint4 q[2]; v16bf v; } u;
  u.q[0] = p[0];      // K = kb .. kb+7
  u.q[1] = p[2];      // K = kb+16 .. kb+23
  return u.v;
}
// B-matrix fragments pre-packed: [kt][nt][lane][16 halves] contiguous.
__device__ inline v16bf load_b_pk(const unsigned short* pk, int ntiles, int kt, int nt, int lane) {
  const v16bf* vp = (const v16bf*)pk;
  return vp[(kt * ntiles + nt) * 32 + lane];
}
__device__ inline v8f wmma_bf16(v16bf a, v16bf b, v8f c) {
  return __builtin_amdgcn_wmma_f32_16x16x32_bf16(false, a, false, b, (short)0, c, false, false);
}
__device__ inline float silu(float x) { return x / (1.0f + __expf(-x)); }

// async copy 16B global -> LDS (tracked by ASYNCcnt)
__device__ inline void async_copy_b128(unsigned lds_off, const void* gaddr) {
  asm volatile("global_load_async_to_lds_b128 %0, %1, off"
               :: "v"(lds_off), "v"((unsigned long long)(size_t)gaddr)
               : "memory");
}
__device__ inline void wait_asynccnt0() {
  asm volatile("s_wait_asynccnt 0x0" ::: "memory");
}

// ---------------------------------------------------------------------------
// Kernel 0: pack fp32 weight W (N_out, K) row-major into bf16 B-fragment tiles.
//   n = nt*16 + (l&15);  kbase = (l<16)?0:8;
//   k = kt*32 + (e<8 ? kbase+e : 16+kbase+(e-8))
// ---------------------------------------------------------------------------
__global__ __launch_bounds__(256) void pack_b_kernel(
    const float* __restrict__ src, unsigned short* __restrict__ dst, int K, int N) {
  int idx = blockIdx.x * blockDim.x + threadIdx.x;
  if (idx >= K * N) return;
  int e    = idx & 15;
  int lane = (idx >> 4) & 31;
  int tile = idx >> 9;
  int ntiles = N >> 4;
  int nt = tile % ntiles;
  int kt = tile / ntiles;
  int n  = nt * 16 + (lane & 15);
  int kb = (lane & 16) ? 8 : 0;
  int k  = kt * 32 + ((e < 8) ? (kb + e) : (16 + kb + (e - 8)));
  dst[idx] = f2bfbits(src[(long)n * K + k]);
}

// ---------------------------------------------------------------------------
// Kernel 1: gather + FM MLP.  Each 256-thread block (8 wave32) owns 32 rows
// (2 M-tiles); B fragments hoisted to registers and reused across M-tiles.
// ---------------------------------------------------------------------------
__global__ __launch_bounds__(256) void anfm_fm_kernel(
    const int* __restrict__ team_ids, const float* __restrict__ emb,
    const unsigned short* __restrict__ w1p, const float* __restrict__ b1,
    const unsigned short* __restrict__ w2p, const float* __restrict__ b2,
    unsigned short* __restrict__ hout) {
  __shared__ __align__(16) unsigned short xs [32 * PD];  // 16 KB
  __shared__ __align__(16) unsigned short h1s[32 * H2];  // 32 KB
  const int tid  = threadIdx.x;
  const long row0 = (long)blockIdx.x * 32;

  // gather 32 embedding rows (float4 loads, packed uint2 stores)
  for (int i = tid; i < 32 * (PD / 4); i += 256) {
    int r = i >> 6, c4 = (i & 63) * 4;
    int pid = team_ids[row0 + r];
    float4 f = *(const float4*)&emb[(long)pid * PD + c4];
    uint2 o;
    o.x = (unsigned)f2bfbits(f.x) | ((unsigned)f2bfbits(f.y) << 16);
    o.y = (unsigned)f2bfbits(f.z) | ((unsigned)f2bfbits(f.w) << 16);
    *(uint2*)&xs[r * PD + c4] = o;
  }
  __syncthreads();

  const int wave = tid >> 5, lane = tid & 31;
  const int n_lane = lane & 15;
  const int mbase  = (lane & 16) ? 8 : 0;

  // GEMM1: h1 = silu(x@W1^T + b1).  32 N-tiles over 8 waves; K = 8 tiles.
  for (int nt = wave; nt < 32; nt += 8) {
    v16bf bfr[8];
#pragma unroll
    for (int kt = 0; kt < 8; ++kt) bfr[kt] = load_b_pk(w1p, 32, kt, nt, lane);
    int n = nt * 16 + n_lane;
    float bias = b1[n];
#pragma unroll
    for (int mt = 0; mt < 2; ++mt) {
      v8f acc = {};
#pragma unroll
      for (int kt = 0; kt < 8; ++kt)
        acc = wmma_bf16(load_a_lds(xs + mt * 16 * PD, PD, kt, lane), bfr[kt], acc);
#pragma unroll
      for (int v = 0; v < 8; ++v)
        h1s[(mt * 16 + mbase + v) * H2 + n] = f2bfbits(silu(acc[v] + bias));
    }
  }
  __syncthreads();

  // GEMM2: h = h1@W2^T + b2.  16 N-tiles over 8 waves; K = 16 tiles (2 halves).
  for (int nt = wave; nt < 16; nt += 8) {
    int n = nt * 16 + n_lane;
    float bias = b2[n];
    v8f acc0 = {}, acc1 = {};
#pragma unroll
    for (int half = 0; half < 2; ++half) {
      v16bf bfr[8];
#pragma unroll
      for (int k8 = 0; k8 < 8; ++k8)
        bfr[k8] = load_b_pk(w2p, 16, half * 8 + k8, nt, lane);
#pragma unroll
      for (int k8 = 0; k8 < 8; ++k8) {
        int kt = half * 8 + k8;
        acc0 = wmma_bf16(load_a_lds(h1s,           H2, kt, lane), bfr[k8], acc0);
        acc1 = wmma_bf16(load_a_lds(h1s + 16 * H2, H2, kt, lane), bfr[k8], acc1);
      }
    }
#pragma unroll
    for (int v = 0; v < 8; ++v) {
      hout[(row0 +      mbase + v) * HD + n] = f2bfbits(acc0[v] + bias);
      hout[(row0 + 16 + mbase + v) * HD + n] = f2bfbits(acc1[v] + bias);
    }
  }
}

// ---------------------------------------------------------------------------
// Kernel 2: pair interactions.  Each block owns 4 batches:
//   20 player rows (padded to 32), 80 pair rows (5 M-tiles).
// ---------------------------------------------------------------------------
__global__ __launch_bounds__(256) void anfm_pair_kernel(
    const unsigned short* __restrict__ h,      // (B*TEAM) x 256 bf16
    const unsigned short* __restrict__ attp,   // packed att_w  (K=256 -> 8 kt, N=256)
    const float* __restrict__ att_b,
    const unsigned short* __restrict__ m1p,    // packed mlp_w1 (K=256 -> 8 kt, N=512)
    const float* __restrict__ mlp_b1,
    const float* __restrict__ mlp_w2,          // (1,512) fp32
    const float* __restrict__ mlp_b2,          // (1,)
    float* __restrict__ out) {
  __shared__ __align__(16) unsigned short hs [32 * HD];   // 16 KB (rows 20..31 zero)
  __shared__ __align__(16) unsigned short tmp[80 * HD];   // 40 KB: hw then prods
  __shared__ float score[80];
  __shared__ float o2[80];
  __shared__ float outacc[4];
  const int tid  = threadIdx.x;
  const int wave = tid >> 5, lane = tid & 31;
  const int b0   = blockIdx.x * 4;
  const long grow0 = (long)b0 * TEAM;
  const int n_lane = lane & 15;
  const int mbase  = (lane & 16) ? 8 : 0;

  // async-copy the 20 real player rows global -> LDS (16B granules)
  for (int i = tid; i < 20 * (HD / 8); i += 256) {      // 640 x b128
    int r = i >> 5, c8 = (i & 31) * 8;
    async_copy_b128((unsigned)(size_t)&hs[r * HD + c8],
                    h + (grow0 + r) * HD + c8);
  }
  // zero the pad rows 20..31
  for (int i = tid; i < 12 * (HD / 8); i += 256) {
    int r = 20 + (i >> 5), c8 = (i & 31) * 8;
    uint4 z = {0u, 0u, 0u, 0u};
    *(uint4*)&hs[r * HD + c8] = z;
  }
  if (tid < 80) o2[tid] = 0.0f;
  if (tid < 4)  outacc[tid] = 0.0f;
  wait_asynccnt0();
  __syncthreads();

  // hw = hs @ att_w^T + att_b : 2 M-tiles x 16 N-tiles, K = 8 tiles -> tmp
  for (int nt = wave; nt < 16; nt += 8) {
    v16bf bfr[8];
#pragma unroll
    for (int kt = 0; kt < 8; ++kt) bfr[kt] = load_b_pk(attp, 16, kt, nt, lane);
    int n = nt * 16 + n_lane;
    float bias = att_b[n];
    v8f acc0 = {}, acc1 = {};
#pragma unroll
    for (int kt = 0; kt < 8; ++kt) {
      acc0 = wmma_bf16(load_a_lds(hs,           HD, kt, lane), bfr[kt], acc0);
      acc1 = wmma_bf16(load_a_lds(hs + 16 * HD, HD, kt, lane), bfr[kt], acc1);
    }
#pragma unroll
    for (int v = 0; v < 8; ++v) {
      tmp[(     mbase + v) * HD + n] = f2bfbits(acc0[v] + bias);
      tmp[(16 + mbase + v) * HD + n] = f2bfbits(acc1[v] + bias);
    }
  }
  __syncthreads();

  // score[p] = dot(hw[a_row], hs[b_row]); one uint4 per operand per lane
  for (int p = wave; p < 80; p += 8) {
    int bi = p / NPAIR, pp = p % NPAIR;
    int i  = pp >> 2, jj = pp & 3;
    int j  = jj + (jj >= i ? 1 : 0);
    int ar = bi * TEAM + i, br = bi * TEAM + j;
    uint4 qa = *(const uint4*)&tmp[ar * HD + lane * 8];
    uint4 qb = *(const uint4*)&hs [br * HD + lane * 8];
    float s = 0.0f;
    s = dot2_acc(qa.x, qb.x, s);
    s = dot2_acc(qa.y, qb.y, s);
    s = dot2_acc(qa.z, qb.z, s);
    s = dot2_acc(qa.w, qb.w, s);
#pragma unroll
    for (int off = 16; off > 0; off >>= 1) s += __shfl_xor(s, off, 32);
    if (lane == 0) score[p] = s;
  }
  __syncthreads();

  // prods[p][c] = a*b  (b128 LDS traffic; overwrites hw region in tmp)
  for (int i = tid; i < 80 * (HD / 8); i += 256) {
    int p = i >> 5, c8 = (i & 31) * 8;
    int bi = p / NPAIR, pp = p % NPAIR;
    int ii = pp >> 2, jj = pp & 3;
    int j  = jj + (jj >= ii ? 1 : 0);
    uint4 qa = *(const uint4*)&hs[(bi * TEAM + ii) * HD + c8];
    uint4 qb = *(const uint4*)&hs[(bi * TEAM + j)  * HD + c8];
    uint4 qo;
    qo.x = pkmul_bf16(qa.x, qb.x);
    qo.y = pkmul_bf16(qa.y, qb.y);
    qo.z = pkmul_bf16(qa.z, qb.z);
    qo.w = pkmul_bf16(qa.w, qb.w);
    *(uint4*)&tmp[p * HD + c8] = qo;
  }
  __syncthreads();

  // m = silu(prods @ mlp_w1^T + b1); o2[row] += sum_n m[row][n]*mlp_w2[n]
  // Jobs: (mt in 0..4) x (ng in 0..7), ng = 4 consecutive N-tiles.
  for (int job = wave; job < 40; job += 8) {
    int mt = job >> 3, ng = job & 7;
    v16bf afr[8];
#pragma unroll
    for (int kt = 0; kt < 8; ++kt)
      afr[kt] = load_a_lds(tmp + mt * 16 * HD, HD, kt, lane);
    float o2acc[8] = {0.f, 0.f, 0.f, 0.f, 0.f, 0.f, 0.f, 0.f};
#pragma unroll
    for (int q = 0; q < 4; ++q) {
      int nt = ng * 4 + q;
      v16bf bfr[8];
#pragma unroll
      for (int kt = 0; kt < 8; ++kt) bfr[kt] = load_b_pk(m1p, 32, kt, nt, lane);
      v8f acc = {};
#pragma unroll
      for (int kt = 0; kt < 8; ++kt) acc = wmma_bf16(afr[kt], bfr[kt], acc);
      int n = nt * 16 + n_lane;
      float bias = mlp_b1[n];
      float w2   = mlp_w2[n];
#pragma unroll
      for (int v = 0; v < 8; ++v)
        o2acc[v] += silu(acc[v] + bias) * w2;
    }
#pragma unroll
    for (int v = 0; v < 8; ++v) {
      float s = o2acc[v];
#pragma unroll
      for (int off = 8; off > 0; off >>= 1) s += __shfl_xor(s, off, 16);
      o2acc[v] = s;
    }
    if (n_lane == 0) {
#pragma unroll
      for (int v = 0; v < 8; ++v)
        atomicAdd(&o2[mt * 16 + mbase + v], o2acc[v]);
    }
  }
  __syncthreads();

  // softmax within each group of (TEAM-1) scores; weighted sum over pairs
  if (tid < 4 * TEAM) {
    int bi = tid / TEAM, g = tid % TEAM;
    int base = bi * NPAIR + g * 4;
    float b2v = mlp_b2[0];
    float mx = score[base];
#pragma unroll
    for (int t = 1; t < 4; ++t) mx = fmaxf(mx, score[base + t]);
    float Z = 0.0f, accv = 0.0f;
#pragma unroll
    for (int t = 0; t < 4; ++t) {
      float e = __expf(score[base + t] - mx);
      Z += e;
      accv += (o2[base + t] + b2v) * e;
    }
    atomicAdd(&outacc[bi], accv / Z);
  }
  __syncthreads();
  if (tid < 4) out[b0 + tid] = outacc[tid];
}

// ---------------------------------------------------------------------------
extern "C" void kernel_launch(void* const* d_in, const int* in_sizes, int n_in,
                              void* d_out, int out_size, void* d_ws, size_t ws_size,
                              hipStream_t stream) {
  const int*   team_ids = (const int*)  d_in[0];
  const float* emb      = (const float*)d_in[1];
  const float* fm_w1    = (const float*)d_in[2];
  const float* fm_b1    = (const float*)d_in[3];
  const float* fm_w2    = (const float*)d_in[4];
  const float* fm_b2    = (const float*)d_in[5];
  const float* att_w    = (const float*)d_in[6];
  const float* att_b    = (const float*)d_in[7];
  const float* mlp_w1   = (const float*)d_in[8];
  const float* mlp_b1   = (const float*)d_in[9];
  const float* mlp_w2   = (const float*)d_in[10];
  const float* mlp_b2   = (const float*)d_in[11];
  float* out = (float*)d_out;

  // workspace layout (bf16 halves); total ~43 MB
  unsigned short* ws   = (unsigned short*)d_ws;
  unsigned short* w1p  = ws;                         // 512*256
  unsigned short* w2p  = w1p  + (size_t)H2 * PD;     // 256*512
  unsigned short* attp = w2p  + (size_t)HD * H2;     // 256*256
  unsigned short* m1p  = attp + (size_t)HD * HD;     // 512*256
  unsigned short* hbuf = m1p  + (size_t)MLP_H * HD;  // 81920*256

  pack_b_kernel<<<(H2 * PD)  / 256, 256, 0, stream>>>(fm_w1,  w1p,  PD, H2);  // K=256,N=512
  pack_b_kernel<<<(HD * H2)  / 256, 256, 0, stream>>>(fm_w2,  w2p,  H2, HD);  // K=512,N=256
  pack_b_kernel<<<(HD * HD)  / 256, 256, 0, stream>>>(att_w,  attp, HD, HD);  // K=256,N=256
  pack_b_kernel<<<(MLP_H*HD) / 256, 256, 0, stream>>>(mlp_w1, m1p,  HD, MLP_H);

  anfm_fm_kernel<<<ROWS_TOTAL / 32, 256, 0, stream>>>(
      team_ids, emb, w1p, fm_b1, w2p, fm_b2, hbuf);

  anfm_pair_kernel<<<B_TOTAL / 4, 256, 0, stream>>>(
      hbuf, attp, att_b, m1p, mlp_b1, mlp_w2, mlp_b2, out);
}
